// MultiHeadAttention_1013612282367
// MI455X (gfx1250) — matrix operand order repaired and checked by reading, not compiled, not run on hardware
//
#include <hip/hip_runtime.h>

#define BB  2
#define SS  2048
#define DD  1024
#define HH  16
#define HDD 64
#define RR  (BB * SS)   // 4096 rows (tokens)

typedef __attribute__((ext_vector_type(16))) __bf16 bf16x16;
typedef __attribute__((ext_vector_type(8)))  float  v8f;

union BF16x16 {
    bf16x16 v;
    __bf16  b[16];
    uint4   q[2];
};

__device__ __forceinline__ v8f wmma_bf16(const BF16x16 &a, const BF16x16 &b, v8f c) {
    return __builtin_amdgcn_wmma_f32_16x16x32_bf16(
        /*neg_a=*/false, a.v, /*neg_b=*/false, b.v,
        /*c_mod=*/(short)0, c, /*reuse_a=*/false, /*reuse_b=*/false);
}

// ---------------------------------------------------------------------------
// Kernel 0a: transpose + convert the 4 weight matrices (f32 [k][n]) into
// bf16 [n][k] via a 32x33 LDS tile (both sides coalesced).
// ---------------------------------------------------------------------------
__global__ void __launch_bounds__(256) wtrans_kernel(
    const float* __restrict__ Wq, const float* __restrict__ Wk,
    const float* __restrict__ Wv, const float* __restrict__ Wo,
    __bf16* __restrict__ Wt) {
    __shared__ float tile[32][33];
    int bid = blockIdx.x;
    int mat = bid >> 10;
    int tk  = (bid >> 5) & 31;
    int tn  = bid & 31;
    const float* W = (mat == 0) ? Wq : (mat == 1) ? Wk : (mat == 2) ? Wv : Wo;
    __bf16* Wtm = Wt + (size_t)mat * DD * DD;
    int tx = threadIdx.x & 31;
    int ty = threadIdx.x >> 5;
#pragma unroll
    for (int i = 0; i < 4; ++i)
        tile[ty + i * 8][tx] = W[(size_t)(tk * 32 + ty + i * 8) * DD + tn * 32 + tx];
    __syncthreads();
#pragma unroll
    for (int i = 0; i < 4; ++i)
        Wtm[(size_t)(tn * 32 + ty + i * 8) * DD + tk * 32 + tx] = (__bf16)tile[tx][ty + i * 8];
}

// ---------------------------------------------------------------------------
// Kernel 0b: convert queries/keys/values (f32) to bf16 row-major, 8 elts/thread.
// ---------------------------------------------------------------------------
__global__ void __launch_bounds__(256) xconvert_kernel(
    const float* __restrict__ Xq, const float* __restrict__ Xk,
    const float* __restrict__ Xv, __bf16* __restrict__ Out) {
    int i = blockIdx.x * blockDim.x + threadIdx.x;     // 3 * 2^19 threads
    int m = i >> 19;
    int r = i & ((1 << 19) - 1);
    const float* X = (m == 0) ? Xq : (m == 1) ? Xk : Xv;
    float4 f0 = reinterpret_cast<const float4*>(X)[(size_t)r * 2 + 0];
    float4 f1 = reinterpret_cast<const float4*>(X)[(size_t)r * 2 + 1];
    union { __bf16 b[8]; uint4 q; } o;
    o.b[0] = (__bf16)f0.x; o.b[1] = (__bf16)f0.y; o.b[2] = (__bf16)f0.z; o.b[3] = (__bf16)f0.w;
    o.b[4] = (__bf16)f1.x; o.b[5] = (__bf16)f1.y; o.b[6] = (__bf16)f1.z; o.b[7] = (__bf16)f1.w;
    reinterpret_cast<uint4*>(Out)[(size_t)m * (1 << 19) + r] = o.q;
}

// ---------------------------------------------------------------------------
// Unified GEMM: A(bf16, RRxDD row-major) @ Wt(bf16 [n][k]) + bias.
// One wave computes a 32x64 block: 2 A fragments x 4 B fragments -> 8 WMMAs
// per 32-wide k-step (B reused 2x, A reused 4x).
// MODE 0: bf16 out [B,H,S,HD]  (Q,K)   -- 64-wide block == exactly one head
// MODE 1: bf16 out [B,H,HD,S]  (V^T)
// MODE 2: f32  out row-major   (final projection into d_out)
// ---------------------------------------------------------------------------
template <int MODE>
__global__ void __launch_bounds__(256) gemm32x64_kernel(
    const __bf16* __restrict__ A, const __bf16* __restrict__ Wt,
    const float* __restrict__ bias, void* __restrict__ Out) {

    int wid  = (blockIdx.x * blockDim.x + threadIdx.x) >> 5;
    int lane = threadIdx.x & 31;
    int lr   = lane & 15;
    int h    = lane >> 4;
    int rt   = wid >> 4;     // 0..127 row block (32 rows)
    int cm   = wid & 15;     // 0..15 macro col block (64 cols == head cm)

    const __bf16* arow0 = A  + (size_t)(rt * 32 + lr) * DD;
    const __bf16* arow1 = arow0 + (size_t)16 * DD;
    const __bf16* wcol  = Wt + (size_t)(cm * 64 + lr) * DD;   // +t*16*DD per sub-tile

    v8f acc[2][4];
#pragma unroll
    for (int r = 0; r < 2; ++r)
#pragma unroll
        for (int t = 0; t < 4; ++t)
            acc[r][t] = v8f{0.f,0.f,0.f,0.f,0.f,0.f,0.f,0.f};

    for (int k = 0; k < DD; k += 32) {
        BF16x16 a[2];   // A: element e <-> K = k + e + 16h (contig 16)
        a[0].q[0] = *reinterpret_cast<const uint4*>(arow0 + k + 16 * h);
        a[0].q[1] = *reinterpret_cast<const uint4*>(arow0 + k + 16 * h + 8);
        a[1].q[0] = *reinterpret_cast<const uint4*>(arow1 + k + 16 * h);
        a[1].q[1] = *reinterpret_cast<const uint4*>(arow1 + k + 16 * h + 8);
        BF16x16 b[4];   // B: element e <-> K = k + e + 16h (contig 16)
#pragma unroll
        for (int t = 0; t < 4; ++t) {
            b[t].q[0] = *reinterpret_cast<const uint4*>(wcol + t * 16 * DD + k + 16 * h);
            b[t].q[1] = *reinterpret_cast<const uint4*>(wcol + t * 16 * DD + k + 16 * h + 8);
        }
#pragma unroll
        for (int r = 0; r < 2; ++r)
#pragma unroll
            for (int t = 0; t < 4; ++t)
                acc[r][t] = wmma_bf16(a[r], b[t], acc[r][t]);
    }

    float bn[4];
#pragma unroll
    for (int t = 0; t < 4; ++t) bn[t] = bias[cm * 64 + t * 16 + lr];

#pragma unroll
    for (int r = 0; r < 2; ++r) {
#pragma unroll
        for (int v = 0; v < 8; ++v) {
            int row = rt * 32 + r * 16 + v + 8 * h;   // token index
            float rv[4];
#pragma unroll
            for (int t = 0; t < 4; ++t) rv[t] = acc[r][t][v] + bn[t];
            if (MODE == 2) {
                float* of = (float*)Out;
                size_t base = (size_t)row * DD + (size_t)cm * 64;
#pragma unroll
                for (int t = 0; t < 4; ++t) of[base + t * 16 + lr] = rv[t];
            } else {
                __bf16* ob = (__bf16*)Out;
                int bidx = row >> 11;
                int s    = row & (SS - 1);
                if (MODE == 0) {
                    size_t base = ((size_t)(bidx * HH + cm) * SS + s) * HDD;
#pragma unroll
                    for (int t = 0; t < 4; ++t) ob[base + t * 16 + lr] = (__bf16)rv[t];
                } else {
                    size_t base = (size_t)(bidx * HH + cm) * HDD * SS + s;
#pragma unroll
                    for (int t = 0; t < 4; ++t) ob[base + (size_t)(t * 16 + lr) * SS] = (__bf16)rv[t];
                }
            }
        }
    }
}

// ---------------------------------------------------------------------------
// Causal flash attention: one wave per (b,h,16-query tile); 64 keys/iter.
// Software-pipelined: batch-load all 8 K fragments -> 8 score WMMAs; issue all
// 8 V-fragment loads BEFORE the softmax so their latency overlaps the VALU
// pass; then LDS P transpose + 8 PV WMMAs on resident registers.
// ---------------------------------------------------------------------------
__global__ void __launch_bounds__(128) attn_kernel(
    const __bf16* __restrict__ Qb, const __bf16* __restrict__ Kb,
    const __bf16* __restrict__ Vt, __bf16* __restrict__ Ctx) {

    __shared__ __align__(16) __bf16 Plds[4][16][64];   // per-wave P tile (8 KB)

    int wlocal = threadIdx.x >> 5;
    int wid    = (blockIdx.x * blockDim.x + threadIdx.x) >> 5;
    int lane   = threadIdx.x & 31;
    int lr     = lane & 15;
    int h      = lane >> 4;
    int qt     = wid & 127;
    int bh     = wid >> 7;

    const __bf16* Qh = Qb + (size_t)bh * SS * HDD;
    const __bf16* Kh = Kb + (size_t)bh * SS * HDD;
    const __bf16* Vh = Vt + (size_t)bh * HDD * SS;

    int q0   = qt * 16;
    int qrow = q0 + lr;

    BF16x16 qa0, qa1;   // Q tile: A fragments for HD k-steps 0 and 32
    qa0.q[0] = *(const uint4*)(Qh + (size_t)qrow * HDD + 16 * h);
    qa0.q[1] = *(const uint4*)(Qh + (size_t)qrow * HDD + 8 + 16 * h);
    qa1.q[0] = *(const uint4*)(Qh + (size_t)qrow * HDD + 32 + 16 * h);
    qa1.q[1] = *(const uint4*)(Qh + (size_t)qrow * HDD + 40 + 16 * h);

    const float SCL = 0.125f * 1.4426950408889634f;   // (1/sqrt(64)) * log2(e)

    float mrow[8], lrow[8];
    v8f o0 = {0.f,0.f,0.f,0.f,0.f,0.f,0.f,0.f};
    v8f o1 = o0, o2 = o0, o3 = o0;
#pragma unroll
    for (int v = 0; v < 8; ++v) { mrow[v] = -3e38f; lrow[v] = 0.f; }

    int nkt = (q0 + 15) / 64 + 1;          // causal: 64-key tiles up to the diagonal
    for (int jt = 0; jt < nkt; ++jt) {
        int kb0 = jt * 64;

        // ---- batch-load all 8 K fragments (distinct registers) ----
        BF16x16 kf[8];                     // [ks2*4 + t]
#pragma unroll
        for (int ks2 = 0; ks2 < 2; ++ks2)
#pragma unroll
            for (int t = 0; t < 4; ++t) {
                const __bf16* kp = Kh + (size_t)(kb0 + t * 16 + lr) * HDD + ks2 * 32 + 16 * h;
                kf[ks2 * 4 + t].q[0] = *(const uint4*)kp;
                kf[ks2 * 4 + t].q[1] = *(const uint4*)(kp + 8);
            }

        // ---- 8 score WMMAs ----
        v8f c[4];
#pragma unroll
        for (int t = 0; t < 4; ++t) c[t] = v8f{0.f,0.f,0.f,0.f,0.f,0.f,0.f,0.f};
#pragma unroll
        for (int ks2 = 0; ks2 < 2; ++ks2)
#pragma unroll
            for (int t = 0; t < 4; ++t)
                c[t] = wmma_bf16(ks2 ? qa1 : qa0, kf[ks2 * 4 + t], c[t]);

        // ---- issue all 8 V fragment loads now; latency hides under softmax ----
        BF16x16 vf[8];                     // [nt*2 + ks2]
#pragma unroll
        for (int nt = 0; nt < 4; ++nt)
#pragma unroll
            for (int ks2 = 0; ks2 < 2; ++ks2) {
                const __bf16* vp = Vh + (size_t)(nt * 16 + lr) * SS + kb0 + ks2 * 32 + 16 * h;
                vf[nt * 2 + ks2].q[0] = *(const uint4*)vp;
                vf[nt * 2 + ks2].q[1] = *(const uint4*)(vp + 8);
            }

        // ---- online softmax over 64 columns ----
#pragma unroll
        for (int v = 0; v < 8; ++v) {
            int q = q0 + v + 8 * h;
            float s[4];
#pragma unroll
            for (int t = 0; t < 4; ++t) {
                s[t] = c[t][v] * SCL;                       // base-2 logits
                if (kb0 + t * 16 + lr > q) s[t] = -1e9f;    // causal mask
            }
            float mx = fmaxf(fmaxf(s[0], s[1]), fmaxf(s[2], s[3]));
#pragma unroll
            for (int d = 1; d < 16; d <<= 1) mx = fmaxf(mx, __shfl_xor(mx, d, 32));
            float mnew = fmaxf(mrow[v], mx);
            float p[4];
#pragma unroll
            for (int t = 0; t < 4; ++t) p[t] = exp2f(s[t] - mnew);
            float rs = (p[0] + p[1]) + (p[2] + p[3]);
#pragma unroll
            for (int d = 1; d < 16; d <<= 1) rs += __shfl_xor(rs, d, 32);
            float corr = exp2f(mrow[v] - mnew);
            lrow[v] = lrow[v] * corr + rs;
            mrow[v] = mnew;
            o0[v] *= corr; o1[v] *= corr; o2[v] *= corr; o3[v] *= corr;
#pragma unroll
            for (int t = 0; t < 4; ++t)
                Plds[wlocal][v + 8 * h][t * 16 + lr] = (__bf16)p[t];
        }
        asm volatile("" ::: "memory");
        BF16x16 pa0, pa1;   // A-layout reload of P (key k-steps 0 and 32)
        pa0.q[0] = *(const uint4*)(&Plds[wlocal][lr][16 * h]);
        pa0.q[1] = *(const uint4*)(&Plds[wlocal][lr][8 + 16 * h]);
        pa1.q[0] = *(const uint4*)(&Plds[wlocal][lr][32 + 16 * h]);
        pa1.q[1] = *(const uint4*)(&Plds[wlocal][lr][40 + 16 * h]);
        asm volatile("" ::: "memory");

        // ---- 8 PV WMMAs on resident V fragments ----
        o0 = wmma_bf16(pa0, vf[0], o0);  o0 = wmma_bf16(pa1, vf[1], o0);
        o1 = wmma_bf16(pa0, vf[2], o1);  o1 = wmma_bf16(pa1, vf[3], o1);
        o2 = wmma_bf16(pa0, vf[4], o2);  o2 = wmma_bf16(pa1, vf[5], o2);
        o3 = wmma_bf16(pa0, vf[6], o3);  o3 = wmma_bf16(pa1, vf[7], o3);
    }

    int bidx = bh >> 4;
    int hh   = bh & 15;
#pragma unroll
    for (int v = 0; v < 8; ++v) {
        int s = q0 + v + 8 * h;
        float inv = 1.0f / lrow[v];
        size_t base = ((size_t)bidx * SS + s) * DD + (size_t)hh * HDD;
        Ctx[base + 0  + lr] = (__bf16)(o0[v] * inv);
        Ctx[base + 16 + lr] = (__bf16)(o1[v] * inv);
        Ctx[base + 32 + lr] = (__bf16)(o2[v] * inv);
        Ctx[base + 48 + lr] = (__bf16)(o3[v] * inv);
    }
}

// ---------------------------------------------------------------------------
extern "C" void kernel_launch(void* const* d_in, const int* in_sizes, int n_in,
                              void* d_out, int out_size, void* d_ws, size_t ws_size,
                              hipStream_t stream) {
    const float* queries = (const float*)d_in[0];
    const float* keys    = (const float*)d_in[1];
    const float* values  = (const float*)d_in[2];
    // d_in[3] = attn_mask (causal, computed analytically in-kernel)
    const float* W_q = (const float*)d_in[4];
    const float* b_q = (const float*)d_in[5];
    const float* W_k = (const float*)d_in[6];
    const float* b_k = (const float*)d_in[7];
    const float* W_v = (const float*)d_in[8];
    const float* b_v = (const float*)d_in[9];
    const float* W_o = (const float*)d_in[10];
    const float* b_o = (const float*)d_in[11];

    char* ws = (char*)d_ws;
    __bf16* Wt  = (__bf16*)(ws);                 // 4 x 2 MiB transposed bf16 weights
    __bf16* Xb  = (__bf16*)(ws + (8u  << 20));   // 3 x 8 MiB bf16 activations (q,k,v)
    __bf16* Qb  = (__bf16*)(ws + (32u << 20));   // [B,H,S,HD]
    __bf16* Kb  = (__bf16*)(ws + (40u << 20));   // [B,H,S,HD]
    __bf16* Vt  = (__bf16*)(ws + (48u << 20));   // [B,H,HD,S]
    __bf16* Ctx = (__bf16*)(ws + (56u << 20));   // [B,S,D]
    (void)ws_size; (void)in_sizes; (void)n_in; (void)out_size;

    wtrans_kernel<<<4096, 256, 0, stream>>>(W_q, W_k, W_v, W_o, Wt);
    xconvert_kernel<<<(3 * (1 << 19)) / 256, 256, 0, stream>>>(queries, keys, values, Xb);

    const size_t XN = (size_t)RR * DD;           // elements per activation tensor
    gemm32x64_kernel<0><<<256, 256, 0, stream>>>(Xb + 0 * XN, Wt + 0 * DD * DD, b_q, Qb);
    gemm32x64_kernel<0><<<256, 256, 0, stream>>>(Xb + 1 * XN, Wt + 1 * DD * DD, b_k, Kb);
    gemm32x64_kernel<1><<<256, 256, 0, stream>>>(Xb + 2 * XN, Wt + 2 * DD * DD, b_v, Vt);

    attn_kernel<<<1024, 128, 0, stream>>>(Qb, Kb, Vt, Ctx);

    gemm32x64_kernel<2><<<256, 256, 0, stream>>>(Ctx, Wt + 3 * DD * DD, b_o, d_out);
}
